// RSSM_68728066671060
// MI455X (gfx1250) — matrix operand order repaired and checked by reading, not compiled, not run on hardware
//
#include <hip/hip_runtime.h>

// ---------------------------------------------------------------------------
// RSSM forward for MI455X (gfx1250, wave32, WMMA bf16, async LDS staging)
// B=8192, A=8, D=4096, H=1024, S=32, C=32, NB=8, BS=512, IBS=256
// out row = [h_new(4096) | prior_logits(1024) | prior_probs(1024) | post_logits(1024)]
// ---------------------------------------------------------------------------

typedef __attribute__((ext_vector_type(16))) __bf16 v16bf;
typedef __attribute__((ext_vector_type(8)))  __bf16 v8bf;
typedef __attribute__((ext_vector_type(8)))  float  v8f;

#define WMMA_BF16(a, b, c) \
  __builtin_amdgcn_wmma_f32_16x16x32_bf16(false, (a), false, (b), (short)0, (c), false, false)

__device__ __forceinline__ __bf16 f2bf(float f) {
  unsigned u = __float_as_uint(f);
  u += 0x7FFFu + ((u >> 16) & 1u);          // round-to-nearest-even
  unsigned short h = (unsigned short)(u >> 16);
  __bf16 r;
  __builtin_memcpy(&r, &h, 2);
  return r;
}

__device__ __forceinline__ v16bf cat8(v8bf lo, v8bf hi) {
  v16bf r;
#pragma unroll
  for (int i = 0; i < 8; ++i) { r[i] = lo[i]; r[i + 8] = hi[i]; }
  return r;
}

__device__ __forceinline__ float sigmoidf_(float x) { return 1.0f / (1.0f + __expf(-x)); }

// CDNA5 async copy: global -> LDS, 16B per lane, tracked by ASYNCcnt.
// lds_off is a per-lane LDS byte address (dynamic-LDS-relative == LDS_BASE-relative).
__device__ __forceinline__ void async_ld_b128(unsigned lds_off, const void* g) {
  asm volatile("global_load_async_to_lds_b128 %0, %1, off"
               :: "v"(lds_off), "v"((unsigned long long)(uintptr_t)g)
               : "memory");
}
__device__ __forceinline__ void wait_async_le3() {
  asm volatile("s_wait_asynccnt 0x3" ::: "memory");
}
__device__ __forceinline__ void wait_async_0() {
  asm volatile("s_wait_asynccnt 0x0" ::: "memory");
}

// ---------------------------------------------------------------------------
// fp32 -> bf16 convert (weights)
// ---------------------------------------------------------------------------
__global__ void cvt_kernel(const float* __restrict__ src, __bf16* __restrict__ dst, long n) {
  for (long i = blockIdx.x * 256ll + threadIdx.x; i < n; i += (long)gridDim.x * 256)
    dst[i] = f2bf(src[i]);
}

// fp32 -> bf16 with per-row is_first zeroing (stoch reset)
__global__ void mask_cvt_kernel(const float* __restrict__ src, const int* __restrict__ is_first,
                                __bf16* __restrict__ dst, long n, int rowlen) {
  for (long i = blockIdx.x * 256ll + threadIdx.x; i < n; i += (long)gridDim.x * 256) {
    long b = i / rowlen;
    float v = is_first[b] ? 0.0f : src[i];
    dst[i] = f2bf(v);
  }
}

// embed -> catbf[:, 4096:5120]
__global__ void embed_cvt_kernel(const float* __restrict__ e, __bf16* __restrict__ catbf) {
  long i = blockIdx.x * 256ll + threadIdx.x;   // over B*1024
  long b = i >> 10;
  int  j = (int)(i & 1023);
  catbf[b * 5120 + 4096 + j] = f2bf(e[i]);
}

// action-normalize + tiny K=8 GEMM:  x[:, 0:1024] = act @ W_act^T + b_act
__global__ void encode_ae_kernel(const float* __restrict__ pa, const float* __restrict__ Wact,
                                 const float* __restrict__ bact, __bf16* __restrict__ xbf) {
  long i = blockIdx.x * 256ll + threadIdx.x;   // over B*1024
  long b = i >> 10;
  int  h = (int)(i & 1023);
  float a[8], mag = 1.0f;
#pragma unroll
  for (int k = 0; k < 8; ++k) { a[k] = pa[b * 8 + k]; mag = fmaxf(mag, fabsf(a[k])); }
  float s = bact[h];
  float inv = 1.0f / mag;
#pragma unroll
  for (int k = 0; k < 8; ++k) s += (a[k] * inv) * Wact[h * 8 + k];
  xbf[b * 2048 + h] = f2bf(s);
}

// ---------------------------------------------------------------------------
// Generic bf16 WMMA GEMM:  D[M,N] = A[M,K] @ W[N,K]^T + bias   (M=8192, N%64==0, K%32==0)
// WG = 256 threads (8 waves). WG tile 128x64; wave tile 32x32 (2x2 WMMA 16x16x32).
// Double-buffered async (global -> LDS) staging: issue stage k+1, then
// s_wait_asynccnt 3 guarantees stage k landed (async loads complete in order).
// Dynamic LDS layout per stage: A[128][40] then B[64][40] bf16.
// ---------------------------------------------------------------------------
#define GEMM_LDSA 5120                      // 128*40 elems
#define GEMM_LDSB 2560                      // 64*40 elems
#define GEMM_STAGE (GEMM_LDSA + GEMM_LDSB)  // 7680 elems = 15360 bytes

template <bool SILU, bool BF16OUT>
__global__ __launch_bounds__(256)
void gemm_bf16_kernel(const __bf16* __restrict__ A, long lda,
                      const __bf16* __restrict__ W,               // [N,K] row-major
                      const float* __restrict__ bias,
                      void* __restrict__ dstv, long ldd, long dst_coloff,
                      int K) {
  extern __shared__ __bf16 smem[];          // 2 stages

  const int tid  = threadIdx.x;
  const int wave = tid >> 5;
  const int lane = tid & 31;
  const int half = lane >> 4;
  const int l16  = lane & 15;
  const int wm   = wave >> 1;    // 0..3
  const int wn   = wave & 1;     // 0..1
  const long m0  = (long)blockIdx.x * 128;
  const long n0  = (long)blockIdx.y * 64;

  v8f acc[2][2] = {};

  auto issue_stage = [&](int k0, int stage) {
    unsigned base = (unsigned)stage * (GEMM_STAGE * 2);
#pragma unroll
    for (int it = 0; it < 2; ++it) {
      int c = tid + it * 256;
      int row = c >> 2, co = (c & 3) * 8;
      async_ld_b128(base + (unsigned)((row * 40 + co) * 2),
                    &A[(m0 + row) * lda + k0 + co]);
    }
    int brow = tid >> 2, bco = (tid & 3) * 8;
    async_ld_b128(base + (unsigned)((GEMM_LDSA + brow * 40 + bco) * 2),
                  &W[(n0 + brow) * (long)K + k0 + bco]);
  };

  const int nk = K >> 5;
  issue_stage(0, 0);

  for (int kk = 0; kk < nk; ++kk) {
    if (kk + 1 < nk) {
      issue_stage((kk + 1) << 5, (kk + 1) & 1);
      wait_async_le3();                     // stage kk landed (in-order completion)
    } else {
      wait_async_0();
    }
    __syncthreads();

    const __bf16* Asb = smem + (kk & 1) * GEMM_STAGE;
    const __bf16* Bsb = Asb + GEMM_LDSA;

    v16bf afr[2], bfr[2];
#pragma unroll
    for (int sm = 0; sm < 2; ++sm) {
      int arow = wm * 32 + sm * 16 + l16;
      afr[sm] = cat8(*(const v8bf*)&Asb[arow * 40 + half * 8],
                     *(const v8bf*)&Asb[arow * 40 + 16 + half * 8]);
    }
#pragma unroll
    for (int sn = 0; sn < 2; ++sn) {
      int brow = wn * 32 + sn * 16 + l16;
      bfr[sn] = cat8(*(const v8bf*)&Bsb[brow * 40 + half * 8],
                     *(const v8bf*)&Bsb[brow * 40 + 16 + half * 8]);
    }
#pragma unroll
    for (int sm = 0; sm < 2; ++sm)
#pragma unroll
      for (int sn = 0; sn < 2; ++sn)
        acc[sm][sn] = WMMA_BF16(afr[sm], bfr[sn], acc[sm][sn]);
    __syncthreads();                        // protect buffer reuse next iteration
  }

  // epilogue
#pragma unroll
  for (int sm = 0; sm < 2; ++sm)
#pragma unroll
    for (int sn = 0; sn < 2; ++sn) {
      long col = n0 + wn * 32 + sn * 16 + l16;
      float bv = bias[col];
#pragma unroll
      for (int r = 0; r < 8; ++r) {
        long row = m0 + wm * 32 + sm * 16 + r + 8 * half;
        float v = acc[sm][sn][r] + bv;
        if (SILU) v = v * sigmoidf_(v);
        if (BF16OUT)
          ((__bf16*)dstv)[row * ldd + dst_coloff + col] = f2bf(v);
        else
          ((float*)dstv)[row * ldd + dst_coloff + col] = v;
      }
    }
}

// ---------------------------------------------------------------------------
// Fused block-diagonal GRU.
// grid = (B/128, NB=8, 512/32=16). Each wave computes the SAME 32x16 output
// region for all three gates (r,u,c) -> 6 accumulators -> fused GRU update.
// K-loop = 768: first 256 from x-block (bf16, async->LDS), last 512 from det
// (fp32, converted+is_first-masked into LDS via VGPRs). W tiles always async.
// Dynamic LDS: A[128][40] then B[96][40] bf16 (single stage).
// ---------------------------------------------------------------------------
#define GRU_LDSA 5120                       // 128*40 elems

__global__ __launch_bounds__(256)
void gru_kernel(const __bf16* __restrict__ xbf,    // [B,2048]
                const __bf16* __restrict__ Wih,    // [8][1536][256] bf16
                const __bf16* __restrict__ Whh,    // [8][1536][512] bf16
                const float* __restrict__ b_ih,    // [8][1536]
                const float* __restrict__ b_hh,    // [8][1536]
                const float* __restrict__ det,     // [B,4096] fp32
                const int* __restrict__ is_first,  // [B]
                float* __restrict__ out,           // [B,7168], cols 0..4096
                __bf16* __restrict__ catbf) {      // [B,5120], cols 0..4096
  extern __shared__ __bf16 smem[];
  __bf16* As = smem;                 // [128][40]
  __bf16* Bs = smem + GRU_LDSA;      // [96][40]

  const int tid  = threadIdx.x;
  const int wave = tid >> 5;
  const int lane = tid & 31;
  const int half = lane >> 4;
  const int l16  = lane & 15;
  const int wm   = wave >> 1;    // 0..3 -> M offset
  const int wn   = wave & 1;     // 0..1 -> N offset within 32-col strip
  const long m0  = (long)blockIdx.x * 128;
  const int  kb  = blockIdx.y;          // GRU block 0..7
  const int  j0  = blockIdx.z * 32;     // column strip within block [0,512)

  v8f acc[2][3] = {};

  for (int k0 = 0; k0 < 768; k0 += 32) {
    // ---- A tile 128x32 ----
    if (k0 < 256) {
      const __bf16* Ap = xbf + (long)kb * 256;
#pragma unroll
      for (int it = 0; it < 2; ++it) {
        int c = tid + it * 256;
        int row = c >> 2, co = (c & 3) * 8;
        async_ld_b128((unsigned)((row * 40 + co) * 2),
                      &Ap[(m0 + row) * 2048 + k0 + co]);
      }
    } else {
      const float* Dp = det + (long)kb * 512 + (k0 - 256);
#pragma unroll
      for (int it = 0; it < 2; ++it) {
        int c = tid + it * 256;
        int row = c >> 2, co = (c & 3) * 8;
        long rg = m0 + row;
        bool fi = is_first[rg] != 0;
        const float* p = &Dp[rg * 4096 + co];
        float4 f0 = *(const float4*)p;
        float4 f1 = *(const float4*)(p + 4);
        float v[8] = {f0.x, f0.y, f0.z, f0.w, f1.x, f1.y, f1.z, f1.w};
#pragma unroll
        for (int i = 0; i < 8; ++i) As[row * 40 + co + i] = f2bf(fi ? 0.0f : v[i]);
      }
    }
    // ---- B tile: 96 rows (3 gates x 32 cols) x 32 K, async ----
    for (int c = tid; c < 384; c += 256) {
      int lr = c >> 2, co = (c & 3) * 8;
      int g = lr >> 5, jj = lr & 31;
      int wrow = g * 512 + j0 + jj;
      const __bf16* src;
      if (k0 < 256) src = Wih + ((long)kb * 1536 + wrow) * 256 + k0 + co;
      else          src = Whh + ((long)kb * 1536 + wrow) * 512 + (k0 - 256) + co;
      async_ld_b128((unsigned)((GRU_LDSA + lr * 40 + co) * 2), src);
    }
    wait_async_0();
    __syncthreads();

    v16bf afr[2], bfr[3];
#pragma unroll
    for (int sm = 0; sm < 2; ++sm) {
      int arow = wm * 32 + sm * 16 + l16;
      afr[sm] = cat8(*(const v8bf*)&As[arow * 40 + half * 8],
                     *(const v8bf*)&As[arow * 40 + 16 + half * 8]);
    }
#pragma unroll
    for (int g = 0; g < 3; ++g) {
      int brow = g * 32 + wn * 16 + l16;
      bfr[g] = cat8(*(const v8bf*)&Bs[brow * 40 + half * 8],
                    *(const v8bf*)&Bs[brow * 40 + 16 + half * 8]);
    }
#pragma unroll
    for (int sm = 0; sm < 2; ++sm)
#pragma unroll
      for (int g = 0; g < 3; ++g)
        acc[sm][g] = WMMA_BF16(afr[sm], bfr[g], acc[sm][g]);
    __syncthreads();
  }

  // ---- fused GRU epilogue ----
  int col = j0 + wn * 16 + l16;                    // [0,512)
  const float* bih = b_ih + (long)kb * 1536;
  const float* bhh = b_hh + (long)kb * 1536;
  float br = bih[col]        + bhh[col];
  float bu = bih[512 + col]  + bhh[512 + col];
  float bc = bih[1024 + col] + bhh[1024 + col];
#pragma unroll
  for (int sm = 0; sm < 2; ++sm) {
#pragma unroll
    for (int r = 0; r < 8; ++r) {
      long row = m0 + wm * 32 + sm * 16 + r + 8 * half;
      float gr = acc[sm][0][r] + br;
      float gu = acc[sm][1][r] + bu;
      float gc = acc[sm][2][r] + bc;
      float rs = sigmoidf_(gr);
      float us = sigmoidf_(gu);
      float cd = tanhf(rs * gc);
      float hb = is_first[row] ? 0.0f : det[row * 4096 + (long)kb * 512 + col];
      float hn = (1.0f - us) * hb + us * cd;
      out[row * 7168 + (long)kb * 512 + col]   = hn;
      catbf[row * 5120 + (long)kb * 512 + col] = f2bf(hn);
    }
  }
}

// ---------------------------------------------------------------------------
// softmax over C=32 per (b,s) + unimix; reads logits strip, writes probs strip
// ---------------------------------------------------------------------------
__global__ void softmax_kernel(float* __restrict__ out) {
  long i = blockIdx.x * 256ll + threadIdx.x;   // over B*1024; 32 lanes per group
  long row = i >> 10;
  int  sc  = (int)(i & 1023);
  float v = out[row * 7168 + 4096 + sc];
  float m = v;
#pragma unroll
  for (int o = 16; o > 0; o >>= 1) m = fmaxf(m, __shfl_xor(m, o, 32));
  float e = __expf(v - m);
  float s = e;
#pragma unroll
  for (int o = 16; o > 0; o >>= 1) s += __shfl_xor(s, o, 32);
  out[row * 7168 + 5120 + sc] = 0.99f * (e / s) + 0.01f / 32.0f;
}

// ---------------------------------------------------------------------------
// launcher
// ---------------------------------------------------------------------------
extern "C" void kernel_launch(void* const* d_in, const int* in_sizes, int n_in,
                              void* d_out, int out_size, void* d_ws, size_t ws_size,
                              hipStream_t stream) {
  (void)in_sizes; (void)n_in; (void)out_size; (void)ws_size;
  const float* det     = (const float*)d_in[0];
  const float* stoch   = (const float*)d_in[1];
  const float* pact    = (const float*)d_in[2];
  const float* embed   = (const float*)d_in[3];
  const int*   isf     = (const int*)d_in[4];
  const float* W_act   = (const float*)d_in[5];
  const float* b_act   = (const float*)d_in[6];
  const float* W_stoch = (const float*)d_in[7];
  const float* b_stoch = (const float*)d_in[8];
  const float* W_ih    = (const float*)d_in[9];
  const float* b_ih    = (const float*)d_in[10];
  const float* W_hh    = (const float*)d_in[11];
  const float* b_hh    = (const float*)d_in[12];
  const float* W_pr1   = (const float*)d_in[13];
  const float* b_pr1   = (const float*)d_in[14];
  const float* W_pr2   = (const float*)d_in[15];
  const float* b_pr2   = (const float*)d_in[16];
  const float* W_po1   = (const float*)d_in[17];
  const float* b_po1   = (const float*)d_in[18];
  const float* W_po2   = (const float*)d_in[19];
  const float* b_po2   = (const float*)d_in[20];
  float* out = (float*)d_out;

  char* ws = (char*)d_ws;
  size_t off = 0;
  auto take = [&](size_t elems) -> __bf16* {
    __bf16* p = (__bf16*)(ws + off);
    off += (elems * 2 + 255) & ~(size_t)255;
    return p;
  };
  __bf16* Ws_bf   = take(1024ull * 1024);        // W_stoch
  __bf16* Wih_bf  = take(8ull * 1536 * 256);
  __bf16* Whh_bf  = take(8ull * 1536 * 512);
  __bf16* Wpr1_bf = take(1024ull * 4096);
  __bf16* Wpr2_bf = take(1024ull * 1024);
  __bf16* Wpo1_bf = take(1024ull * 5120);
  __bf16* Wpo2_bf = take(1024ull * 1024);
  __bf16* xbf     = take(8192ull * 2048);        // [ae | se]
  __bf16* stochbf = take(8192ull * 1024);        // masked stoch
  __bf16* catbf   = take(8192ull * 5120);        // [h_new | embed]
  __bf16* phbf    = take(8192ull * 1024);        // silu hidden (prior then posterior)

  // ---- weight conversions (stream-ordered, graph-capturable) ----
  cvt_kernel<<<2048, 256, 0, stream>>>(W_stoch, Ws_bf,   1024ll * 1024);
  cvt_kernel<<<2048, 256, 0, stream>>>(W_ih,    Wih_bf,  8ll * 1536 * 256);
  cvt_kernel<<<2048, 256, 0, stream>>>(W_hh,    Whh_bf,  8ll * 1536 * 512);
  cvt_kernel<<<2048, 256, 0, stream>>>(W_pr1,   Wpr1_bf, 1024ll * 4096);
  cvt_kernel<<<2048, 256, 0, stream>>>(W_pr2,   Wpr2_bf, 1024ll * 1024);
  cvt_kernel<<<2048, 256, 0, stream>>>(W_po1,   Wpo1_bf, 1024ll * 5120);
  cvt_kernel<<<2048, 256, 0, stream>>>(W_po2,   Wpo2_bf, 1024ll * 1024);

  // ---- activation prep ----
  mask_cvt_kernel<<<4096, 256, 0, stream>>>(stoch, isf, stochbf, 8192ll * 1024, 1024);
  encode_ae_kernel<<<32768, 256, 0, stream>>>(pact, W_act, b_act, xbf);
  embed_cvt_kernel<<<32768, 256, 0, stream>>>(embed, catbf);

  dim3 blk(256);
  const size_t gemm_lds = 2 * GEMM_STAGE * sizeof(__bf16);   // 30720 B
  const size_t gru_lds  = (GRU_LDSA + 96 * 40) * sizeof(__bf16); // 17920 B

  // se = stoch @ W_stoch^T + b  -> xbf[:, 1024:2048]
  gemm_bf16_kernel<false, true><<<dim3(64, 16), blk, gemm_lds, stream>>>(
      stochbf, 1024, Ws_bf, b_stoch, (void*)xbf, 2048, 1024, 1024);

  // fused block-diagonal GRU -> out[:, :4096] (fp32) and catbf[:, :4096] (bf16)
  gru_kernel<<<dim3(64, 8, 16), blk, gru_lds, stream>>>(
      xbf, Wih_bf, Whh_bf, b_ih, b_hh, det, isf, out, catbf);

  // prior: ph = silu(h @ Wpr1^T + b)  ;  logits = ph @ Wpr2^T + b -> out[:, 4096:5120]
  gemm_bf16_kernel<true, true><<<dim3(64, 16), blk, gemm_lds, stream>>>(
      catbf, 5120, Wpr1_bf, b_pr1, (void*)phbf, 1024, 0, 4096);
  gemm_bf16_kernel<false, false><<<dim3(64, 16), blk, gemm_lds, stream>>>(
      phbf, 1024, Wpr2_bf, b_pr2, (void*)out, 7168, 4096, 1024);

  // prior probs (unimix softmax) -> out[:, 5120:6144]
  softmax_kernel<<<32768, 256, 0, stream>>>(out);

  // posterior: qh = silu([h|embed] @ Wpo1^T + b) ; logits -> out[:, 6144:7168]
  gemm_bf16_kernel<true, true><<<dim3(64, 16), blk, gemm_lds, stream>>>(
      catbf, 5120, Wpo1_bf, b_po1, (void*)phbf, 1024, 0, 5120);
  gemm_bf16_kernel<false, false><<<dim3(64, 16), blk, gemm_lds, stream>>>(
      phbf, 1024, Wpo2_bf, b_po2, (void*)out, 7168, 6144, 1024);
}
